// SlideFFN_69965017252079
// MI455X (gfx1250) — matrix operand order repaired and checked by reading, not compiled, not run on hardware
//
#include <hip/hip_runtime.h>
#include <hip/hip_bf16.h>
#include <stdint.h>

// ---------------- problem constants ----------------
#define DIM   1024
#define NN    4096
#define LL    4
#define KK    8
#define TT    256
#define SS    1024
#define CC    32
#define BT    8192           // total tokens
#define MARGIN_F 0.1f

typedef __attribute__((ext_vector_type(16))) __bf16 v16bf;
typedef __attribute__((ext_vector_type(8)))  float  v8f;
typedef __attribute__((ext_vector_type(4)))  unsigned int u32x4;

union FragB16 { u32x4 q[2]; v16bf v; };

__device__ __forceinline__ uint16_t f2bf(float f) {
    uint32_t u = __float_as_uint(f);
    uint32_t r = u + 0x7FFFu + ((u >> 16) & 1u);   // round-to-nearest-even
    return (uint16_t)(r >> 16);
}

// ---------------- workspace layout (bytes) ----------------
// xn fp32 [8192][1024], xbt bf16 [1024][8192] (X transposed),
// lshbf bf16 [4096][1024], w2bf bf16, act bf16 [32][256][1024],
// tcode u32[8192], ncode u32[4096], scores i32[32][4096],
// ids i32[32][1024], posdot/negdot f32[32]
static constexpr size_t OFF_XN     = 0;
static constexpr size_t OFF_XBT    = OFF_XN     + (size_t)BT * DIM * 4;
static constexpr size_t OFF_LSHBF  = OFF_XBT    + (size_t)BT * DIM * 2;
static constexpr size_t OFF_W2BF   = OFF_LSHBF  + (size_t)NN * DIM * 2;
static constexpr size_t OFF_ACT    = OFF_W2BF   + (size_t)NN * DIM * 2;
static constexpr size_t OFF_TCODE  = OFF_ACT    + (size_t)CC * TT * SS * 2;
static constexpr size_t OFF_NCODE  = OFF_TCODE  + (size_t)BT * 4;
static constexpr size_t OFF_SCORES = OFF_NCODE  + (size_t)NN * 4;
static constexpr size_t OFF_IDS    = OFF_SCORES + (size_t)CC * NN * 4;
static constexpr size_t OFF_POS    = OFF_IDS    + (size_t)CC * SS * 4;
static constexpr size_t OFF_NEG    = OFF_POS    + 128;
static constexpr size_t WS_NEED    = OFF_NEG    + 128;

// ---------------- 1) convert weights to bf16 ----------------
__global__ __launch_bounds__(256) void cvt_k(const float* __restrict__ a,
                                             const float* __restrict__ b,
                                             uint16_t* __restrict__ abf,
                                             uint16_t* __restrict__ bbf) {
    size_t base = (size_t)blockIdx.x * 1024 + threadIdx.x;
    #pragma unroll
    for (int j = 0; j < 4; ++j) {
        size_t i = base + (size_t)j * 256;
        abf[i] = f2bf(a[i]);
        bbf[i] = f2bf(b[i]);
    }
}

// ---------------- 2) LayerNorm ----------------
__global__ __launch_bounds__(256) void ln_k(const float* __restrict__ hs,
                                            const float* __restrict__ gma,
                                            const float* __restrict__ bta,
                                            float* __restrict__ xn) {
    __shared__ float red[256];
    const int t = blockIdx.x, tid = threadIdx.x;
    const float* x = hs + (size_t)t * DIM;
    float v[4]; float s = 0.f;
    #pragma unroll
    for (int j = 0; j < 4; ++j) { v[j] = x[tid + j * 256]; s += v[j]; }
    red[tid] = s; __syncthreads();
    for (int off = 128; off > 0; off >>= 1) {
        if (tid < off) red[tid] += red[tid + off];
        __syncthreads();
    }
    const float mu = red[0] * (1.0f / DIM);
    __syncthreads();
    float s2 = 0.f;
    #pragma unroll
    for (int j = 0; j < 4; ++j) { float d = v[j] - mu; s2 += d * d; }
    red[tid] = s2; __syncthreads();
    for (int off = 128; off > 0; off >>= 1) {
        if (tid < off) red[tid] += red[tid + off];
        __syncthreads();
    }
    const float inv = rsqrtf(red[0] * (1.0f / DIM) + 1e-12f);
    #pragma unroll
    for (int j = 0; j < 4; ++j) {
        int idx = tid + j * 256;
        xn[(size_t)t * DIM + idx] = (v[j] - mu) * inv * gma[idx] + bta[idx];
    }
}

// ---------------- 2b) tiled transpose: xn fp32 [t][d] -> xbt bf16 [d][t] ------
__global__ __launch_bounds__(256) void xpose_k(const float* __restrict__ xn,
                                               uint16_t* __restrict__ xbt) {
    __shared__ uint16_t tile[32][33];
    const int bid = blockIdx.x;                  // (8192/32) * (1024/32) = 8192
    const int tb = bid >> 5, db = bid & 31;      // token block, dim block
    const int tid = threadIdx.x;
    const int r = tid >> 5, cl = tid & 31;       // 8 rows x 32 cols per pass
    #pragma unroll
    for (int j = 0; j < 4; ++j) {
        int row = r + j * 8;
        tile[row][cl] = f2bf(xn[(size_t)(tb * 32 + row) * DIM + db * 32 + cl]);
    }
    __syncthreads();
    #pragma unroll
    for (int j = 0; j < 4; ++j) {
        int row = r + j * 8;                     // d offset within block
        xbt[(size_t)(db * 32 + row) * BT + tb * 32 + cl] = tile[cl][row];
    }
}

// ---------------- 3) simhash codes (tokens + neurons), 1 wave/row ----------------
__global__ __launch_bounds__(256) void hash_k(const float* __restrict__ xn,
                                              const float* __restrict__ lshw,
                                              const float* __restrict__ hw,
                                              unsigned int* __restrict__ tcode,
                                              unsigned int* __restrict__ ncode) {
    const int wv   = (blockIdx.x << 3) + (threadIdx.x >> 5);
    const int lane = threadIdx.x & 31;
    const float* src; unsigned int* dst;
    if (wv < BT) { src = xn + (size_t)wv * DIM;            dst = tcode + wv; }
    else         { int r = wv - BT;
                   src = lshw + (size_t)r * DIM;           dst = ncode + r; }
    const float* h = hw + (size_t)lane * DIM;      // lane = l*8+k projection
    float acc = 0.f;
    for (int d = 0; d < DIM; ++d) acc = fmaf(src[d], h[d], acc);
    unsigned int mask = __builtin_amdgcn_ballot_w32(acc > 0.0f);
    if (lane == 0) *dst = mask;                    // byte l of mask == code of table l
}

// ---------------- 4) collision scores via histograms, 1 block/chunk ----------------
__global__ __launch_bounds__(256) void score_k(const unsigned int* __restrict__ tcode,
                                               const unsigned int* __restrict__ ncode,
                                               int* __restrict__ scores) {
    __shared__ int hist[LL * 256];
    const int c = blockIdx.x, tid = threadIdx.x;
    #pragma unroll
    for (int j = 0; j < LL; ++j) hist[tid + j * 256] = 0;
    __syncthreads();
    unsigned int code = tcode[c * TT + tid];
    #pragma unroll
    for (int l = 0; l < LL; ++l)
        atomicAdd(&hist[l * 256 + ((code >> (l * 8)) & 255u)], 1);
    __syncthreads();
    for (int i = 0; i < NN / 256; ++i) {
        int n = tid + i * 256;
        unsigned int nc = ncode[n];
        int s = 0;
        #pragma unroll
        for (int l = 0; l < LL; ++l)
            s += hist[l * 256 + ((nc >> (l * 8)) & 255u)];
        scores[(size_t)c * NN + n] = s;
    }
}

// ---------------- 5) counting-select top-S, 1 block/chunk ----------------
__global__ __launch_bounds__(256) void topk_k(const int* __restrict__ scores,
                                              int* __restrict__ ids) {
    __shared__ int hist[TT * LL + 1];          // scores in [0, 1024]
    __shared__ int pa[256], pt[256];
    __shared__ int sthr, srem, sabove;
    const int c = blockIdx.x, tid = threadIdx.x;
    const int* sc = scores + (size_t)c * NN;
    for (int i = tid; i < TT * LL + 1; i += 256) hist[i] = 0;
    __syncthreads();
    const int base = tid * 16;                 // contiguous range -> index order
    for (int i = 0; i < 16; ++i) atomicAdd(&hist[sc[base + i]], 1);
    __syncthreads();
    if (tid == 0) {
        int cum = 0;
        for (int v = TT * LL; v >= 0; --v) {
            cum += hist[v];
            if (cum >= SS) {
                sthr = v; sabove = cum - hist[v]; srem = SS - sabove;
                break;
            }
        }
    }
    __syncthreads();
    const int thr = sthr, rem = srem, aboveTot = sabove;
    int ca = 0, ct = 0;
    for (int i = 0; i < 16; ++i) {
        int s = sc[base + i];
        ca += (s > thr); ct += (s == thr);
    }
    pa[tid] = ca; pt[tid] = ct; __syncthreads();
    if (tid == 0) {                            // serial exclusive scans (256)
        int aA = 0, aT = 0;
        for (int i = 0; i < 256; ++i) {
            int ta = pa[i]; pa[i] = aA; aA += ta;
            int tt = pt[i]; pt[i] = aT; aT += tt;
        }
    }
    __syncthreads();
    int aPos = pa[tid], tPos = pt[tid];
    int* oid = ids + (size_t)c * SS;
    for (int i = 0; i < 16; ++i) {
        int n = base + i, s = sc[n];
        if (s > thr) oid[aPos++] = n;
        else if (s == thr) { if (tPos < rem) oid[aboveTot + tPos] = n; tPos++; }
    }
}

// ---------------- 6) triplet-loss statistics (factorized means) ----------------
__global__ __launch_bounds__(256) void cvec_k(const float* __restrict__ xn,
                                              const float* __restrict__ lshw,
                                              const int* __restrict__ ids,
                                              float* __restrict__ posdot,
                                              float* __restrict__ negdot) {
    __shared__ float redp[256], redn[256];
    const int c = blockIdx.x, tid = threadIdx.x;
    float xs[4] = {0,0,0,0}, wp[4] = {0,0,0,0}, wn_[4] = {0,0,0,0};
    for (int t = 0; t < TT; ++t) {
        const float* r = xn + (size_t)(c * TT + t) * DIM;
        #pragma unroll
        for (int j = 0; j < 4; ++j) xs[j] += r[tid + j * 256];
    }
    for (int s = 0; s < SS; ++s) {
        int id  = ids[(size_t)c * SS + s];
        int nid = (id + NN / 2) & (NN - 1);
        const float* a  = lshw + (size_t)id  * DIM;
        const float* b2 = lshw + (size_t)nid * DIM;
        #pragma unroll
        for (int j = 0; j < 4; ++j) {
            wp[j]  += a[tid + j * 256];
            wn_[j] += b2[tid + j * 256];
        }
    }
    float p = 0.f, n = 0.f;
    #pragma unroll
    for (int j = 0; j < 4; ++j) { p += xs[j] * wp[j]; n += xs[j] * wn_[j]; }
    redp[tid] = p; redn[tid] = n; __syncthreads();
    for (int off = 128; off > 0; off >>= 1) {
        if (tid < off) { redp[tid] += redp[tid + off]; redn[tid] += redn[tid + off]; }
        __syncthreads();
    }
    if (tid == 0) { posdot[c] = redp[0]; negdot[c] = redn[0]; }
}

__global__ __launch_bounds__(32) void loss_k(const float* __restrict__ posdot,
                                             const float* __restrict__ negdot,
                                             float* __restrict__ out, int lossIdx) {
    __shared__ float r[32];
    const int tid = threadIdx.x;
    const float inv = 1.0f / ((float)TT * (float)SS);
    r[tid] = fmaxf(0.0f, MARGIN_F + (negdot[tid] - posdot[tid]) * inv);
    __syncthreads();
    if (tid == 0) {
        float s = 0.f;
        for (int i = 0; i < 32; ++i) s += r[i];
        out[lossIdx] = s * (1.0f / 32.0f);
    }
}

// ---------------- 7) GEMM1 (transposed orientation):
// tile D[s][t] = (gathered W1 rows) x X^T  -> exact GELU -> act[t][s] (bf16)
// block = 256 thr (8 waves); block tile 128(M=s) x 128(N=t); K=1024 steps of 32
// A staged in LDS row-major (no transpose needed); B read directly from xbt.
__global__ __launch_bounds__(256) void gemm1_k(const uint16_t* __restrict__ xbt,
                                               const uint16_t* __restrict__ w1bf,
                                               const int* __restrict__ ids,
                                               uint16_t* __restrict__ act) {
    __shared__ __align__(16) uint16_t ldsA[128 * 32];    // [s][k] row-major
    const int tid = threadIdx.x, lane = tid & 31, wave = tid >> 5;
    const int bid = blockIdx.x;
    const int c  = bid >> 4, sb = (bid >> 1) & 7, tb = bid & 1;
    const int sBase   = sb * 128;                        // s base within chunk
    const int tokLoc  = tb * 128;                        // t base within chunk

    // A loader: 2 chunks of 8 bf16 per thread (gathered W1 rows)
    const int cj0 = tid * 2, cj1 = tid * 2 + 1;
    const int ar0 = cj0 >> 2, ac0 = (cj0 & 3) * 8;
    const int ar1 = cj1 >> 2, ac1 = (cj1 & 3) * 8;
    const int sid0 = ids[(size_t)c * SS + sBase + ar0];
    const int sid1 = ids[(size_t)c * SS + sBase + ar1];
    const uint16_t* aSrc0 = w1bf + (size_t)sid0 * DIM + ac0;
    const uint16_t* aSrc1 = w1bf + (size_t)sid1 * DIM + ac1;

    const int wm = wave >> 2, wn = wave & 3;             // wave tile: 64(M=s) x 32(N=t)
    const int hlf = lane >> 4, m16 = lane & 15;
    // B source: xbt[d][token]; lane supplies N column (= token m16), 16 K rows (d) in table order
    const uint16_t* bBase = xbt + (size_t)c * TT + tokLoc + wn * 32 + m16;

    v8f acc[4][2];
    #pragma unroll
    for (int i = 0; i < 4; ++i)
        #pragma unroll
        for (int j = 0; j < 2; ++j)
            #pragma unroll
            for (int r = 0; r < 8; ++r) acc[i][j][r] = 0.0f;

    for (int kb = 0; kb < DIM; kb += 32) {
        *(u32x4*)&ldsA[ar0 * 32 + ac0] = *(const u32x4*)(aSrc0 + kb);
        *(u32x4*)&ldsA[ar1 * 32 + ac1] = *(const u32x4*)(aSrc1 + kb);
        FragB16 bF[2];
        #pragma unroll
        for (int nt = 0; nt < 2; ++nt) {     // direct global, strided in d (K)
            const uint16_t* bp = bBase + (size_t)kb * BT + nt * 16;
            #pragma unroll
            for (int s = 0; s < 16; ++s) {
                const int k = (s < 8) ? hlf * 8 + s : 16 + hlf * 8 + (s - 8);
                ((uint16_t*)&bF[nt])[s] = bp[(size_t)k * BT];
            }
        }
        __syncthreads();
        #pragma unroll
        for (int mt = 0; mt < 4; ++mt) {
            const int mrow = wm * 64 + mt * 16 + m16;
            FragB16 aF;
            aF.q[0] = *(const u32x4*)&ldsA[mrow * 32 + hlf * 8];
            aF.q[1] = *(const u32x4*)&ldsA[mrow * 32 + 16 + hlf * 8];
            #pragma unroll
            for (int nt = 0; nt < 2; ++nt)
                acc[mt][nt] = __builtin_amdgcn_wmma_f32_16x16x32_bf16(
                    false, aF.v, false, bF[nt].v, (short)0, acc[mt][nt], false, false);
        }
        __syncthreads();
    }
    // epilogue: exact GELU, store into act[t][s] (row = token, col = s)
    #pragma unroll
    for (int mt = 0; mt < 4; ++mt)
        #pragma unroll
        for (int nt = 0; nt < 2; ++nt) {
            const int tcol = tokLoc + wn * 32 + nt * 16 + m16;       // local t
            #pragma unroll
            for (int r = 0; r < 8; ++r) {
                const int srow = sBase + wm * 64 + mt * 16 + r + 8 * hlf; // s
                float f = acc[mt][nt][r];
                float g = 0.5f * f * (1.0f + erff(f * 0.70710678118f));
                act[(size_t)(c * TT + tcol) * SS + srow] = f2bf(g);
            }
        }
}

// ---------------- 8) GEMM2: out = act * W2[ids] + bias (fp32 store) ----------
__global__ __launch_bounds__(256) void gemm2_k(const uint16_t* __restrict__ act,
                                               const uint16_t* __restrict__ w2bf,
                                               const int* __restrict__ ids,
                                               const float* __restrict__ bias,
                                               float* __restrict__ out) {
    __shared__ __align__(16) uint16_t ldsA[128 * 32];
    const int tid = threadIdx.x, lane = tid & 31, wave = tid >> 5;
    const int bid = blockIdx.x;
    const int c  = bid >> 4, mb = (bid >> 3) & 1, nb = bid & 7;
    const int tokBase = c * TT + mb * 128;
    const int dBase   = nb * 128;

    const int cj0 = tid * 2, cj1 = tid * 2 + 1;
    const int ar0 = cj0 >> 2, ac0 = (cj0 & 3) * 8;
    const int ar1 = cj1 >> 2, ac1 = (cj1 & 3) * 8;
    const uint16_t* aSrc0 = act + (size_t)(tokBase + ar0) * SS + ac0;
    const uint16_t* aSrc1 = act + (size_t)(tokBase + ar1) * SS + ac1;

    const int wm = wave >> 2, wn = wave & 3;
    const int hlf = lane >> 4, m16 = lane & 15;

    v8f acc[4][2];
    #pragma unroll
    for (int i = 0; i < 4; ++i)
        #pragma unroll
        for (int j = 0; j < 2; ++j)
            #pragma unroll
            for (int r = 0; r < 8; ++r) acc[i][j][r] = 0.0f;

    for (int kb = 0; kb < SS; kb += 32) {
        *(u32x4*)&ldsA[ar0 * 32 + ac0] = *(const u32x4*)(aSrc0 + kb);
        *(u32x4*)&ldsA[ar1 * 32 + ac1] = *(const u32x4*)(aSrc1 + kb);
        const int sid = ids[(size_t)c * SS + kb + lane];     // B row for lane (K=lane)
        __syncthreads();
        FragB16 bF[2];
        #pragma unroll
        for (int nt = 0; nt < 2; ++nt) {
            const uint16_t* brow = w2bf + dBase + wn * 32 + nt * 16 + m16;
            #pragma unroll
            for (int s = 0; s < 16; ++s) {           // lane = N column, elements over K
                const int k = (s < 8) ? hlf * 8 + s : 16 + hlf * 8 + (s - 8);
                const int rk = __shfl(sid, k);
                ((uint16_t*)&bF[nt])[s] = brow[(size_t)rk * DIM];
            }
        }
        #pragma unroll
        for (int mt = 0; mt < 4; ++mt) {
            const int mrow = wm * 64 + mt * 16 + m16;
            FragB16 aF;
            aF.q[0] = *(const u32x4*)&ldsA[mrow * 32 + hlf * 8];
            aF.q[1] = *(const u32x4*)&ldsA[mrow * 32 + 16 + hlf * 8];
            #pragma unroll
            for (int nt = 0; nt < 2; ++nt)
                acc[mt][nt] = __builtin_amdgcn_wmma_f32_16x16x32_bf16(
                    false, aF.v, false, bF[nt].v, (short)0, acc[mt][nt], false, false);
        }
        __syncthreads();
    }
    #pragma unroll
    for (int mt = 0; mt < 4; ++mt)
        #pragma unroll
        for (int nt = 0; nt < 2; ++nt) {
            const int col = dBase + wn * 32 + nt * 16 + m16;
            const float bz = bias[col];
            #pragma unroll
            for (int r = 0; r < 8; ++r) {
                const int trow = tokBase + wm * 64 + mt * 16 + r + 8 * hlf;
                out[(size_t)trow * DIM + col] = acc[mt][nt][r] + bz;
            }
        }
}

// ---------------- launcher ----------------
extern "C" void kernel_launch(void* const* d_in, const int* in_sizes, int n_in,
                              void* d_out, int out_size, void* d_ws, size_t ws_size,
                              hipStream_t stream) {
    if (ws_size < WS_NEED) return;
    const float* hs    = (const float*)d_in[0];
    const float* gma   = (const float*)d_in[1];
    const float* bta   = (const float*)d_in[2];
    const float* lshw  = (const float*)d_in[3];
    const float* wts   = (const float*)d_in[4];
    const float* bias  = (const float*)d_in[5];
    const float* hw    = (const float*)d_in[6];
    float* out = (float*)d_out;

    char* ws = (char*)d_ws;
    float*        xn     = (float*)(ws + OFF_XN);
    uint16_t*     xbt    = (uint16_t*)(ws + OFF_XBT);
    uint16_t*     lshbf  = (uint16_t*)(ws + OFF_LSHBF);
    uint16_t*     w2bf   = (uint16_t*)(ws + OFF_W2BF);
    uint16_t*     actb   = (uint16_t*)(ws + OFF_ACT);
    unsigned int* tcode  = (unsigned int*)(ws + OFF_TCODE);
    unsigned int* ncode  = (unsigned int*)(ws + OFF_NCODE);
    int*          scores = (int*)(ws + OFF_SCORES);
    int*          ids    = (int*)(ws + OFF_IDS);
    float*        posd   = (float*)(ws + OFF_POS);
    float*        negd   = (float*)(ws + OFF_NEG);

    cvt_k  <<<NN * DIM / 1024, 256, 0, stream>>>(lshw, wts, lshbf, w2bf);
    ln_k   <<<BT, 256, 0, stream>>>(hs, gma, bta, xn);
    xpose_k<<<(BT / 32) * (DIM / 32), 256, 0, stream>>>(xn, xbt);
    hash_k <<<(BT + NN) / 8, 256, 0, stream>>>(xn, lshw, hw, tcode, ncode);
    score_k<<<CC, 256, 0, stream>>>(tcode, ncode, scores);
    topk_k <<<CC, 256, 0, stream>>>(scores, ids);
    cvec_k <<<CC, 256, 0, stream>>>(xn, lshw, ids, posd, negd);
    loss_k <<<1, 32, 0, stream>>>(posd, negd, out, out_size - 1);
    gemm1_k<<<CC * 16, 256, 0, stream>>>(xbt, lshbf, ids, actb);
    gemm2_k<<<CC * 16, 256, 0, stream>>>(actb, w2bf, ids, bias, out);
}
